// MMSingleStreamBlockReFuser_15324443312838
// MI455X (gfx1250) — matrix operand order repaired and checked by reading, not compiled, not run on hardware
//
#include <hip/hip_runtime.h>
#include <math.h>

// ---------------- problem constants (B=1) ----------------
#define S_LEN 2304
#define TXT_L 256
#define IMG_L 2048       // S - TXT
#define HID   3072
#define NH    24
#define HD    128
#define MLP   12288
#define RL    512
#define QKV3  (3*HID)      // 9216
#define L1OUT (3*HID+MLP)  // 21504
#define CAT   (HID+MLP)    // 15360
#define ATTN_SCALE 0.08838834764831845f  // HD^-0.5

typedef __bf16         bf16x16 __attribute__((ext_vector_type(16)));
typedef __bf16         bf16x2  __attribute__((ext_vector_type(2)));
typedef float          f32x8   __attribute__((ext_vector_type(8)));
typedef float          f32x4   __attribute__((ext_vector_type(4)));
typedef unsigned int   u32x4   __attribute__((ext_vector_type(4)));
typedef int            i32x4   __attribute__((ext_vector_type(4)));
typedef int            i32x8   __attribute__((ext_vector_type(8)));
typedef unsigned short u16;

union Frag { bf16x16 bv; u32x4 q[2]; };

#if defined(__has_builtin)
#if __has_builtin(__builtin_amdgcn_tensor_load_to_lds) && \
    __has_builtin(__builtin_amdgcn_s_wait_tensorcnt)
#define HAVE_TDM 1
#endif
#if __has_builtin(__builtin_amdgcn_cvt_pk_bf16_f32)
#define HAVE_PKBF 1
#endif
#endif

__device__ __forceinline__ u16 f2bf(float f) {
  unsigned int u = __builtin_bit_cast(unsigned int, f);
  u = (u + 0x7fffu + ((u >> 16) & 1u)) >> 16;
  return (u16)u;
}
__device__ __forceinline__ unsigned pack2(float a, float b) {
#if defined(HAVE_PKBF)
  bf16x2 r = __builtin_amdgcn_cvt_pk_bf16_f32(a, b);
  return __builtin_bit_cast(unsigned, r);
#else
  return (unsigned)f2bf(a) | ((unsigned)f2bf(b) << 16);
#endif
}
__device__ __forceinline__ float gelu_tanh(float v) {
  return 0.5f * v * (1.0f + tanhf(0.7978845608028654f * (v + 0.044715f * v * v * v)));
}

#if defined(HAVE_TDM)
// TDM: DMA a 2D bf16 tile [tile_m rows x tile_k cols] of an [M x K] row-major
// tensor (elements of 2 bytes, row stride K elements) into LDS at lds_off.
__device__ __forceinline__ void tdm_load_tile_bf16(const u16* gsrc,
                                                   unsigned lds_off, int tile_k,
                                                   int tile_m, int K, int M) {
  unsigned long long ga = (unsigned long long)(uintptr_t)gsrc;
  u32x4 g0;
  g0[0] = 1u;                                      // count=1 (valid descriptor)
  g0[1] = lds_off;                                 // lds_addr
  g0[2] = (unsigned)(ga & 0xffffffffu);            // global_addr[31:0]
  g0[3] = (unsigned)((ga >> 32) & 0x1ffffffu) |    // global_addr[56:32]
          (2u << 30);                              // type=2 ("image")
  i32x8 g1;
  g1[0] = 0x10000;                                 // data_size=1 (2 bytes)
  g1[1] = (int)((unsigned)(K & 0xffff) << 16);     // tensor_dim0 lo16 @ bit48
  g1[2] = (int)(((unsigned)K >> 16) |              // tensor_dim0 hi16
                ((unsigned)(M & 0xffff) << 16));   // tensor_dim1 lo16
  g1[3] = (int)(((unsigned)M >> 16) |              // tensor_dim1 hi16
                ((unsigned)tile_k << 16));         // tile_dim0
  g1[4] = (int)(unsigned)tile_m;                   // tile_dim1 (tile_dim2=0)
  g1[5] = (int)K;                                  // tensor_dim0_stride lo32
  g1[6] = 0;
  g1[7] = 0;
  i32x4 z4 = {0, 0, 0, 0};
#if __clang_major__ >= 23
  i32x8 z8 = {0, 0, 0, 0, 0, 0, 0, 0};
  __builtin_amdgcn_tensor_load_to_lds(g0, g1, z4, z4, z8, 0);
#else
  __builtin_amdgcn_tensor_load_to_lds(g0, g1, z4, z4, 0);
#endif
}
#endif

// ================= mod = silu(vec) @ mod_w + mod_b =================
__global__ __launch_bounds__(256) void mod_gemv(
    const float* __restrict__ vec, const float* __restrict__ W,
    const float* __restrict__ b, float* __restrict__ out) {
  __shared__ float sv[HID];
  const int t = threadIdx.x;
  for (int k = t; k < HID; k += 256) {
    float v = vec[k];
    sv[k] = v / (1.0f + __expf(-v));
  }
  __syncthreads();
  const int j = blockIdx.x * 256 + t;
  float acc = 0.0f;
  for (int k = 0; k < HID; ++k) acc += sv[k] * W[(size_t)k * QKV3 + j];
  out[j] = acc + b[j];
}

// ============ x_mod = LN(x)*(1+scale)+shift  -> bf16 ============
__global__ __launch_bounds__(256) void ln_mod(
    const float* __restrict__ x, const float* __restrict__ modv,
    u16* __restrict__ xmod) {
  const int row = blockIdx.x;
  const int t = threadIdx.x;
  const int lane = t & 31, wid = t >> 5;
  __shared__ float red[8];
  const float* xr = x + (size_t)row * HID;

  float s = 0.0f;
  for (int c = t; c < HID; c += 256) s += xr[c];
  for (int m = 16; m; m >>= 1) s += __shfl_xor(s, m, 32);
  if (lane == 0) red[wid] = s;
  __syncthreads();
  float tot = 0.0f;
  for (int j = 0; j < 8; ++j) tot += red[j];
  const float mean = tot * (1.0f / HID);
  __syncthreads();

  float v = 0.0f;
  for (int c = t; c < HID; c += 256) { float d = xr[c] - mean; v += d * d; }
  for (int m = 16; m; m >>= 1) v += __shfl_xor(v, m, 32);
  if (lane == 0) red[wid] = v;
  __syncthreads();
  float vt = 0.0f;
  for (int j = 0; j < 8; ++j) vt += red[j];
  const float rstd = rsqrtf(vt * (1.0f / HID) + 1e-6f);

  const float* shift = modv;
  const float* scale = modv + HID;
  for (int c = t; c < HID; c += 256) {
    float val = (xr[c] - mean) * rstd * (1.0f + scale[c]) + shift[c];
    xmod[(size_t)row * HID + c] = f2bf(val);
  }
}

// ================= generic bf16 WMMA GEMM (double-buffered) =================
// C[M][N] = A_bf16[M][K] @ B_f32[K][N] + bias, tile 128x128x32, 256 thr (8 waves).
// A tile staged by the Tensor Data Mover when available; B tile converted
// fp32->bf16 in registers (v_cvt_pk_bf16_f32) and stored transposed [n][k].
// mode 0: store f32 to out_f (ldc)
// mode 1: lin1 split: n<9216 -> qkv f32 ; else gelu -> bf16 into a2[:, 3072+..]
// mode 2: final: out = x + (acc+bias)*gate
__global__ __launch_bounds__(256) void gemm_bf16(
    const u16* __restrict__ A, const float* __restrict__ B,
    const float* __restrict__ bias, int M, int N, int K, int mode,
    float* __restrict__ out_f, int ldc,
    float* __restrict__ qkv_out, u16* __restrict__ a2_out,
    const float* __restrict__ xres, const float* __restrict__ gate,
    float* __restrict__ out_final) {
  __shared__ u16 sA[2][128 * 32];
  __shared__ u16 sB[2][128 * 32];  // transposed: [n][k]
  const int t = threadIdx.x;
  const int lane = t & 31;
  const int wave = t >> 5;
  const int wm = wave >> 1;  // 0..3
  const int wn = wave & 1;   // 0..1
  const int l15 = lane & 15, lhi = lane >> 4;
  const int bm = blockIdx.y * 128;
  const int bn = blockIdx.x * 128;

  // per-thread staging coordinates
  const int arow = t >> 2;            // A: rows (two passes of 64)
  const int aseg = (t & 3) * 8;       // A: 8 bf16 per 16B
  int bk[2], bn4[2];                  // B: 2 (k-pair) x 4 (n) blocks, 2 iters
#pragma unroll
  for (int i = 0; i < 2; ++i) {
    const int lin = i * 256 + t;
    bk[i] = (lin >> 5) * 2;
    bn4[i] = (lin & 31) * 4;
  }

  f32x8 acc[2][4] = {};

  // ---- prologue: stage tile 0 into buffer 0 ----
#if defined(HAVE_TDM)
  if (wave == 0)
    tdm_load_tile_bf16(A + (size_t)bm * K, (unsigned)(uintptr_t)&sA[0][0], 32,
                       128, K, M);
#else
#pragma unroll
  for (int p = 0; p < 2; ++p) {
    const int row = p * 64 + arow;
    *reinterpret_cast<u32x4*>(&sA[0][row * 32 + aseg]) =
        *reinterpret_cast<const u32x4*>(&A[(size_t)(bm + row) * K + aseg]);
  }
#endif
#pragma unroll
  for (int i = 0; i < 2; ++i) {
    f32x4 r0 = *reinterpret_cast<const f32x4*>(&B[(size_t)bk[i] * N + bn + bn4[i]]);
    f32x4 r1 = *reinterpret_cast<const f32x4*>(&B[(size_t)(bk[i] + 1) * N + bn + bn4[i]]);
#pragma unroll
    for (int j = 0; j < 4; ++j)
      *reinterpret_cast<unsigned*>(&sB[0][(bn4[i] + j) * 32 + bk[i]]) =
          pack2(r0[j], r1[j]);
  }
#if defined(HAVE_TDM)
  if (wave == 0) __builtin_amdgcn_s_wait_tensorcnt(0);
#endif
  __syncthreads();

  const int nk = K / 32;
  for (int ki = 0; ki < nk; ++ki) {
    const int cur = ki & 1, nxt = cur ^ 1;
    const int k0 = ki * 32;
    const bool has_next = (ki + 1) < nk;

    // ---- issue next-tile loads before compute ----
    u32x4 aNext[2];
    f32x4 bNext[2][2];
    if (has_next) {
#if defined(HAVE_TDM)
      if (wave == 0)
        tdm_load_tile_bf16(A + (size_t)bm * K + k0 + 32,
                           (unsigned)(uintptr_t)&sA[nxt][0], 32, 128, K, M);
#else
#pragma unroll
      for (int p = 0; p < 2; ++p) {
        const int row = p * 64 + arow;
        aNext[p] = *reinterpret_cast<const u32x4*>(
            &A[(size_t)(bm + row) * K + k0 + 32 + aseg]);
      }
#endif
#pragma unroll
      for (int i = 0; i < 2; ++i) {
        bNext[i][0] = *reinterpret_cast<const f32x4*>(
            &B[(size_t)(k0 + 32 + bk[i]) * N + bn + bn4[i]]);
        bNext[i][1] = *reinterpret_cast<const f32x4*>(
            &B[(size_t)(k0 + 32 + bk[i] + 1) * N + bn + bn4[i]]);
      }
      if (k0 + 64 < K)
        __builtin_prefetch(&B[(size_t)(k0 + 64 + (t >> 3)) * N + bn + (t & 7) * 16], 0, 3);
    }

    // ---- fragments + WMMA from current buffer ----
    Frag a[2], b[4];
#pragma unroll
    for (int mf = 0; mf < 2; ++mf) {
      const int row = wm * 32 + mf * 16 + l15;
      const int kb = lhi * 8;
      a[mf].q[0] = *reinterpret_cast<const u32x4*>(&sA[cur][row * 32 + kb]);
      a[mf].q[1] = *reinterpret_cast<const u32x4*>(&sA[cur][row * 32 + 16 + kb]);
    }
#pragma unroll
    for (int nf = 0; nf < 4; ++nf) {
      const int n = wn * 64 + nf * 16 + l15;
      const int kb = lhi * 8;
      b[nf].q[0] = *reinterpret_cast<const u32x4*>(&sB[cur][n * 32 + kb]);
      b[nf].q[1] = *reinterpret_cast<const u32x4*>(&sB[cur][n * 32 + 16 + kb]);
    }
#pragma unroll
    for (int mf = 0; mf < 2; ++mf)
#pragma unroll
      for (int nf = 0; nf < 4; ++nf)
        acc[mf][nf] = __builtin_amdgcn_wmma_f32_16x16x32_bf16(
            false, a[mf].bv, false, b[nf].bv, (short)0, acc[mf][nf], false, false);

    // ---- publish next buffer ----
    if (has_next) {
#if !defined(HAVE_TDM)
#pragma unroll
      for (int p = 0; p < 2; ++p) {
        const int row = p * 64 + arow;
        *reinterpret_cast<u32x4*>(&sA[nxt][row * 32 + aseg]) = aNext[p];
      }
#endif
#pragma unroll
      for (int i = 0; i < 2; ++i)
#pragma unroll
        for (int j = 0; j < 4; ++j)
          *reinterpret_cast<unsigned*>(&sB[nxt][(bn4[i] + j) * 32 + bk[i]]) =
              pack2(bNext[i][0][j], bNext[i][1][j]);
#if defined(HAVE_TDM)
      if (wave == 0) __builtin_amdgcn_s_wait_tensorcnt(0);
#endif
      __syncthreads();
    }
    (void)aNext;
  }

  // ---- epilogue ----
#pragma unroll
  for (int mf = 0; mf < 2; ++mf)
#pragma unroll
    for (int nf = 0; nf < 4; ++nf) {
      const int n = bn + wn * 64 + nf * 16 + l15;
      const float bb = bias[n];
#pragma unroll
      for (int r = 0; r < 8; ++r) {
        const int m = bm + wm * 32 + mf * 16 + lhi * 8 + r;
        const float v = acc[mf][nf][r] + bb;
        if (mode == 0) {
          out_f[(size_t)m * ldc + n] = v;
        } else if (mode == 1) {
          if (n < QKV3) qkv_out[(size_t)m * QKV3 + n] = v;
          else a2_out[(size_t)m * CAT + HID + (n - QKV3)] = f2bf(gelu_tanh(v));
        } else {
          out_final[(size_t)m * HID + n] =
              xres[(size_t)m * HID + n] + v * gate[n];
        }
      }
    }
}

// ============ qkv post: RMS + RoPE, reorder to head-major bf16 ============
__global__ __launch_bounds__(128) void qkv_post(
    const float* __restrict__ qkv, const float* __restrict__ qs,
    const float* __restrict__ ks, const float* __restrict__ fcos,
    const float* __restrict__ fsin, u16* __restrict__ Qb,
    u16* __restrict__ Kb, u16* __restrict__ VbT) {
  const int s = blockIdx.x, h = blockIdx.y, d = threadIdx.x;
  const int lane = d & 31, wid = d >> 5;
  __shared__ float red[4];
  __shared__ float vecb[HD];
  const float* base = qkv + (size_t)s * QKV3 + h * HD + d;
  const float qv = base[0];
  const float kv = base[HID];
  const float vv = base[2 * HID];
  const bool rope = (s < IMG_L);
  const float c = rope ? fcos[s * HD + d] : 0.0f;
  const float sn = rope ? fsin[s * HD + d] : 0.0f;

  // ---- Q ----
  float t2 = qv * qv;
  for (int m = 16; m; m >>= 1) t2 += __shfl_xor(t2, m, 32);
  if (lane == 0) red[wid] = t2;
  __syncthreads();
  float ms = (red[0] + red[1] + red[2] + red[3]) * (1.0f / HD);
  float qn = qv * rsqrtf(ms + 1e-6f) * qs[d];
  __syncthreads();
  vecb[d] = qn;
  __syncthreads();
  float qo = qn;
  if (rope) {
    const float rot = (d & 1) ? vecb[d - 1] : -vecb[d + 1];
    qo = qn * c + rot * sn;
  }
  Qb[((size_t)h * S_LEN + s) * HD + d] = f2bf(qo);
  __syncthreads();

  // ---- K ----
  t2 = kv * kv;
  for (int m = 16; m; m >>= 1) t2 += __shfl_xor(t2, m, 32);
  if (lane == 0) red[wid] = t2;
  __syncthreads();
  ms = (red[0] + red[1] + red[2] + red[3]) * (1.0f / HD);
  float kn = kv * rsqrtf(ms + 1e-6f) * ks[d];
  __syncthreads();
  vecb[d] = kn;
  __syncthreads();
  float ko = kn;
  if (rope) {
    const float rot = (d & 1) ? vecb[d - 1] : -vecb[d + 1];
    ko = kn * c + rot * sn;
  }
  Kb[((size_t)h * S_LEN + s) * HD + d] = f2bf(ko);

  // ---- V (transposed) ----
  VbT[((size_t)h * HD + d) * S_LEN + s] = f2bf(vv);
}

// ============ ref-latent post: RMS + RoPE (all positions) ============
__global__ __launch_bounds__(128) void ref_post(
    const float* __restrict__ krt, const float* __restrict__ vrt,
    const float* __restrict__ kns, const float* __restrict__ rcos,
    const float* __restrict__ rsin, u16* __restrict__ Krb,
    u16* __restrict__ VrbT) {
  const int s = blockIdx.x, h = blockIdx.y, d = threadIdx.x;
  const int lane = d & 31, wid = d >> 5;
  __shared__ float red[4];
  __shared__ float vecb[HD];
  const float kv = krt[(size_t)s * HID + h * HD + d];
  const float vv = vrt[(size_t)s * HID + h * HD + d];

  float t2 = kv * kv;
  for (int m = 16; m; m >>= 1) t2 += __shfl_xor(t2, m, 32);
  if (lane == 0) red[wid] = t2;
  __syncthreads();
  const float ms = (red[0] + red[1] + red[2] + red[3]) * (1.0f / HD);
  const float kn = kv * rsqrtf(ms + 1e-6f) * kns[d];
  vecb[d] = kn;
  __syncthreads();
  const float rot = (d & 1) ? vecb[d - 1] : -vecb[d + 1];
  const float ko = kn * rcos[s * HD + d] + rot * rsin[s * HD + d];
  Krb[((size_t)h * RL + s) * HD + d] = f2bf(ko);
  VrbT[((size_t)h * HD + d) * RL + s] = f2bf(vv);
}

// ============ fp32 -> bf16 convert ============
__global__ __launch_bounds__(256) void cvt_bf(const float* __restrict__ in,
                                              u16* __restrict__ out, int n) {
  const int i = blockIdx.x * 256 + threadIdx.x;
  if (i < n) out[i] = f2bf(in[i]);
}

// ============ exact-softmax attention, scores in LDS ============
// grid: (S/32, NH), 256 threads. L = kv length. Scores fp32 in dynamic LDS.
// mode 0: attnTmp[s][h*HD+d] = o
// mode 1: a2[s][h*HD+d] = bf16(attnTmp + o)
__global__ __launch_bounds__(256) void attn_kernel(
    const u16* __restrict__ Qb, const u16* __restrict__ Kb,
    const u16* __restrict__ VbT, int L, int mode,
    float* __restrict__ attnTmp, u16* __restrict__ a2_out) {
  extern __shared__ float smem[];
  float* sS = smem;                    // [32][L]
  float* red = smem + 32 * L;          // [32*8]
  float* rowstat = red + 256;          // [64]: 0..31 max, 32..63 inv-sum
  const int t = threadIdx.x;
  const int lane = t & 31;
  const int wave = t >> 5;
  const int l15 = lane & 15, lhi = lane >> 4;
  const int qb = blockIdx.x;  // 32-row q block
  const int h = blockIdx.y;

  // ---- phase 1: scores = Q K^T * scale ----
  Frag aq[2][4];
#pragma unroll
  for (int mf = 0; mf < 2; ++mf)
#pragma unroll
    for (int kk = 0; kk < 4; ++kk) {
      const u16* qp = Qb + ((size_t)h * S_LEN + qb * 32 + mf * 16 + l15) * HD +
                      kk * 32 + lhi * 8;
      aq[mf][kk].q[0] = *reinterpret_cast<const u32x4*>(qp);
      aq[mf][kk].q[1] = *reinterpret_cast<const u32x4*>(qp + 16);
    }
  for (int nc = wave * 16; nc < L; nc += 128) {
    Frag bk[4];
#pragma unroll
    for (int kk = 0; kk < 4; ++kk) {
      const u16* kp = Kb + ((size_t)h * L + nc + l15) * HD + kk * 32 + lhi * 8;
      bk[kk].q[0] = *reinterpret_cast<const u32x4*>(kp);
      bk[kk].q[1] = *reinterpret_cast<const u32x4*>(kp + 16);
    }
    f32x8 sc[2] = {};
#pragma unroll
    for (int kk = 0; kk < 4; ++kk)
#pragma unroll
      for (int mf = 0; mf < 2; ++mf)
        sc[mf] = __builtin_amdgcn_wmma_f32_16x16x32_bf16(
            false, aq[mf][kk].bv, false, bk[kk].bv, (short)0, sc[mf], false, false);
#pragma unroll
    for (int mf = 0; mf < 2; ++mf)
#pragma unroll
      for (int r = 0; r < 8; ++r)
        sS[(mf * 16 + lhi * 8 + r) * L + nc + l15] = sc[mf][r] * ATTN_SCALE;
  }
  __syncthreads();

  // ---- phase 2: softmax (unnormalized exp + inv-sum) ----
  {
    const int row = t >> 3, sub = t & 7;
    float mx = -1e30f;
    for (int n = sub; n < L; n += 8) mx = fmaxf(mx, sS[row * L + n]);
    red[row * 8 + sub] = mx;
  }
  __syncthreads();
  if (t < 32) {
    float mx = -1e30f;
    for (int j = 0; j < 8; ++j) mx = fmaxf(mx, red[t * 8 + j]);
    rowstat[t] = mx;
  }
  __syncthreads();
  {
    const int row = t >> 3, sub = t & 7;
    const float mx = rowstat[row];
    float s = 0.0f;
    for (int n = sub; n < L; n += 8) {
      const float e = __expf(sS[row * L + n] - mx);
      sS[row * L + n] = e;
      s += e;
    }
    red[row * 8 + sub] = s;
  }
  __syncthreads();
  if (t < 32) {
    float s = 0.0f;
    for (int j = 0; j < 8; ++j) s += red[t * 8 + j];
    rowstat[32 + t] = 1.0f / s;
  }
  __syncthreads();

  // ---- phase 3: O = P V (wave owns 16 output dims) ----
  f32x8 oacc[2] = {};
  for (int kv = 0; kv < L; kv += 32) {
    Frag ap[2];
#pragma unroll
    for (int mf = 0; mf < 2; ++mf) {
      const float* pp = sS + (mf * 16 + l15) * L + kv;
      const int kb = lhi * 8;
      f32x4 c0 = *reinterpret_cast<const f32x4*>(pp + kb);
      f32x4 c1 = *reinterpret_cast<const f32x4*>(pp + kb + 4);
      f32x4 c2 = *reinterpret_cast<const f32x4*>(pp + 16 + kb);
      f32x4 c3 = *reinterpret_cast<const f32x4*>(pp + 20 + kb);
      ap[mf].q[0] = (u32x4){pack2(c0[0], c0[1]), pack2(c0[2], c0[3]),
                            pack2(c1[0], c1[1]), pack2(c1[2], c1[3])};
      ap[mf].q[1] = (u32x4){pack2(c2[0], c2[1]), pack2(c2[2], c2[3]),
                            pack2(c3[0], c3[1]), pack2(c3[2], c3[3])};
    }
    Frag bv;
    const u16* vp = VbT + ((size_t)h * HD + wave * 16 + l15) * L + kv + lhi * 8;
    bv.q[0] = *reinterpret_cast<const u32x4*>(vp);
    bv.q[1] = *reinterpret_cast<const u32x4*>(vp + 16);
#pragma unroll
    for (int mf = 0; mf < 2; ++mf)
      oacc[mf] = __builtin_amdgcn_wmma_f32_16x16x32_bf16(
          false, ap[mf].bv, false, bv.bv, (short)0, oacc[mf], false, false);
  }

#pragma unroll
  for (int mf = 0; mf < 2; ++mf)
#pragma unroll
    for (int r = 0; r < 8; ++r) {
      const int m = mf * 16 + lhi * 8 + r;
      const int srow = qb * 32 + m;
      const int col = h * HD + wave * 16 + l15;
      const float o = oacc[mf][r] * rowstat[32 + m];
      if (mode == 0) attnTmp[(size_t)srow * HID + col] = o;
      else a2_out[(size_t)srow * CAT + col] =
               f2bf(attnTmp[(size_t)srow * HID + col] + o);
    }
}

// ======================= host launcher =======================
extern "C" void kernel_launch(void* const* d_in, const int* in_sizes, int n_in,
                              void* d_out, int out_size, void* d_ws,
                              size_t ws_size, hipStream_t stream) {
  (void)in_sizes; (void)n_in; (void)out_size; (void)ws_size;

  const float* x        = (const float*)d_in[0];
  const float* vec      = (const float*)d_in[1];
  const float* fcos     = (const float*)d_in[7];
  const float* fsin     = (const float*)d_in[8];
  const float* ref_lat  = (const float*)d_in[9];
  const float* rcos     = (const float*)d_in[10];
  const float* rsin     = (const float*)d_in[11];
  const float* mod_w    = (const float*)d_in[12];
  const float* mod_b    = (const float*)d_in[13];
  const float* lin1_w   = (const float*)d_in[14];
  const float* lin1_b   = (const float*)d_in[15];
  const float* lin2_w   = (const float*)d_in[16];
  const float* lin2_b   = (const float*)d_in[17];
  const float* q_scale  = (const float*)d_in[18];
  const float* k_scale  = (const float*)d_in[19];
  const float* ref_kw   = (const float*)d_in[20];
  const float* ref_kb   = (const float*)d_in[21];
  const float* ref_vw   = (const float*)d_in[22];
  const float* ref_vb   = (const float*)d_in[23];
  const float* ref_kns  = (const float*)d_in[24];
  float* out = (float*)d_out;

  size_t off = 0;
  auto take = [&](size_t bytes) -> void* {
    void* p = (char*)d_ws + off;
    off += (bytes + 255) & ~(size_t)255;
    return p;
  };
  float* modv   = (float*)take((size_t)QKV3 * 4);
  u16*   xmod   = (u16*)  take((size_t)S_LEN * HID * 2);
  float* qkv    = (float*)take((size_t)S_LEN * QKV3 * 4);
  u16*   A2     = (u16*)  take((size_t)S_LEN * CAT * 2);
  u16*   Qb     = (u16*)  take((size_t)NH * S_LEN * HD * 2);
  u16*   Kb     = (u16*)  take((size_t)NH * S_LEN * HD * 2);
  u16*   VbT    = (u16*)  take((size_t)NH * S_LEN * HD * 2);
  float* attnT  = (float*)take((size_t)S_LEN * HID * 4);
  u16*   refbf  = (u16*)  take((size_t)RL * HID * 2);
  float* krtmp  = (float*)take((size_t)RL * HID * 4);
  float* vrtmp  = (float*)take((size_t)RL * HID * 4);
  u16*   Krb    = (u16*)  take((size_t)NH * RL * HD * 2);
  u16*   VrbT   = (u16*)  take((size_t)NH * RL * HD * 2);

  const size_t shm_main = (size_t)(32 * S_LEN + 256 + 64) * sizeof(float);
  const size_t shm_ref  = (size_t)(32 * RL + 256 + 64) * sizeof(float);
  (void)hipFuncSetAttribute((const void*)attn_kernel,
                            hipFuncAttributeMaxDynamicSharedMemorySize,
                            (int)shm_main);

  // 1. modulation
  mod_gemv<<<QKV3 / 256, 256, 0, stream>>>(vec, mod_w, mod_b, modv);
  // 2. LN + mod -> bf16
  ln_mod<<<S_LEN, 256, 0, stream>>>(x, modv, xmod);
  // 3. lin1 GEMM (split epilogue)
  gemm_bf16<<<dim3(L1OUT / 128, S_LEN / 128), 256, 0, stream>>>(
      xmod, lin1_w, lin1_b, S_LEN, L1OUT, HID, 1,
      nullptr, 0, qkv, A2, nullptr, nullptr, nullptr);
  // 4. qkv post-process
  qkv_post<<<dim3(S_LEN, NH), 128, 0, stream>>>(
      qkv, q_scale, k_scale, fcos, fsin, Qb, Kb, VbT);
  // 5. ref latent projections
  cvt_bf<<<(RL * HID + 255) / 256, 256, 0, stream>>>(ref_lat, refbf, RL * HID);
  gemm_bf16<<<dim3(HID / 128, RL / 128), 256, 0, stream>>>(
      refbf, ref_kw, ref_kb, RL, HID, HID, 0,
      krtmp, HID, nullptr, nullptr, nullptr, nullptr, nullptr);
  gemm_bf16<<<dim3(HID / 128, RL / 128), 256, 0, stream>>>(
      refbf, ref_vw, ref_vb, RL, HID, HID, 0,
      vrtmp, HID, nullptr, nullptr, nullptr, nullptr, nullptr);
  ref_post<<<dim3(RL, NH), 128, 0, stream>>>(
      krtmp, vrtmp, ref_kns, rcos, rsin, Krb, VrbT);
  // 6. attention (main, then +ref)
  attn_kernel<<<dim3(S_LEN / 32, NH), 256, shm_main, stream>>>(
      Qb, Kb, VbT, S_LEN, 0, attnT, nullptr);
  attn_kernel<<<dim3(S_LEN / 32, NH), 256, shm_ref, stream>>>(
      Qb, Krb, VrbT, RL, 1, attnT, A2);
  // 7. lin2 + gated residual -> d_out
  gemm_bf16<<<dim3(HID / 128, S_LEN / 128), 256, 0, stream>>>(
      A2, lin2_w, lin2_b, S_LEN, HID, CAT, 2,
      nullptr, 0, nullptr, nullptr, x, modv + 2 * HID, out);
}
